// VQWorldModel_8804682957163
// MI455X (gfx1250) — matrix operand order repaired and checked by reading, not compiled, not run on hardware
//
#include <hip/hip_runtime.h>
#include <hip/hip_bf16.h>
#include <math.h>

// ---------- CDNA5 WMMA types ----------
typedef __attribute__((ext_vector_type(16))) _Float16 v16h;
typedef __attribute__((ext_vector_type(8)))  _Float16 v8h;
typedef __attribute__((ext_vector_type(8)))  float    v8f;

#define WMMA_F16(a,b,c) __builtin_amdgcn_wmma_f32_16x16x32_f16(false,(a),false,(b),(short)0,(c),false,false)

// =====================================================================
// Elementwise / packing kernels
// =====================================================================
__global__ void cvt16_k(const float* __restrict__ s, _Float16* __restrict__ d, int n){
    int i = blockIdx.x*blockDim.x + threadIdx.x;
    if (i < n) d[i] = (_Float16)s[i];
}

// forward conv weights: OIHW fp32 -> [co][K] f16, K zero-padded to Kp (mult of 32)
__global__ void pack_fwd_w_k(const float* __restrict__ w, _Float16* __restrict__ dst,
                             int Cout, int Kin, int Kp){
    int i = blockIdx.x*blockDim.x + threadIdx.x;
    if (i >= Cout*Kp) return;
    int co = i / Kp, k = i - co*Kp;
    dst[i] = (k < Kin) ? (_Float16)w[(size_t)co*Kin + k] : (_Float16)0;
}

// transposed-conv weights: per parity class pc=(py*2+px) pack w[co][ci][2ty+py][2tx+px]
// into dst[pc][co][ci*4 + ty*2 + tx], Cout zero-padded to Cp
__global__ void pack_t_w_k(const float* __restrict__ w, _Float16* __restrict__ dst,
                           int Cout, int Cin, int Cp){
    int Kt = Cin*4;
    int i = blockIdx.x*blockDim.x + threadIdx.x;
    if (i >= 4*Cp*Kt) return;
    int pc  = i / (Cp*Kt);
    int rem = i - pc*(Cp*Kt);
    int co  = rem / Kt;
    int k   = rem - co*Kt;
    int ci = k >> 2, tt = k & 3, ty = tt >> 1, tx = tt & 1;
    int py = pc >> 1, px = pc & 1;
    _Float16 v = (_Float16)0;
    if (co < Cout)
        v = (_Float16)w[(((size_t)co*Cin + ci)*4 + (2*ty+py))*4 + (2*tx+px)];
    dst[i] = v;
}

__global__ void init_k(int* __restrict__ hist, float* __restrict__ scal){
    int i = blockIdx.x*blockDim.x + threadIdx.x;
    if (i < 8192) hist[i] = 0;
    if (i < 64)   scal[i] = 0.f;
}

// =====================================================================
// Implicit-GEMM forward conv (k=4,s=2,p=1), WMMA f32<-f16
//   M=16 out channels, N=16 out pixels (one output row segment), K=Cin2*16
//   Cin2 = Cin rounded up to even; channels >= Cin staged as zeros so the
//   inner loop is branch-free (loads + WMMA only).
// mode: 0 = bias only (z layer), 1 = bias + affine + relu
// =====================================================================
__global__ void conv_fwd_k(const _Float16* __restrict__ in, const _Float16* __restrict__ wpk,
                           const float* __restrict__ cb, const float* __restrict__ sc,
                           const float* __restrict__ ab,
                           _Float16* __restrict__ out16, float* __restrict__ out32,
                           int B, int Cin, int H, int W, int Cout, int Kp, int mode){
    extern __shared__ _Float16 smem[];               // [Cin2][4][36] (34 cols used)
    const int OH = H >> 1, OW = W >> 1, tpr = OW >> 4;
    const int Cin2 = (Cin + 1) & ~1;
    int t   = blockIdx.x;
    int jx  = t % tpr;
    int oy  = (t / tpr) % OH;
    int nb  = t / (tpr * OH);
    int ox0 = jx << 4;
    (void)B;

    // stage input slab: rows 2*oy-1..2*oy+2, cols 2*ox0-1..2*ox0+32 (34 cols)
    int total = Cin2 * 4 * 34;
    for (int e = threadIdx.x; e < total; e += blockDim.x){
        int ci = e / 136; int rem = e - ci*136;
        int ry = rem / 34; int cx = rem - ry*34;
        int iy = 2*oy - 1 + ry, ix = 2*ox0 - 1 + cx;
        _Float16 v = (_Float16)0;
        if (ci < Cin && iy >= 0 && iy < H && ix >= 0 && ix < W)
            v = in[(((size_t)nb*Cin + ci)*H + iy)*W + ix];
        smem[(ci*4 + ry)*36 + cx] = v;
    }
    __syncthreads();

    int lane = threadIdx.x & 31, wv = threadIdx.x >> 5, nW = blockDim.x >> 5;
    int hi = lane >> 4, nl = lane & 15;
    int chunks = Kp >> 5;

    for (int ct = wv; ct < (Cout >> 4); ct += nW){
        v8f acc = {};
        const _Float16* wrow = wpk + (size_t)(ct*16 + nl)*Kp + hi*16;
        const _Float16* bbase = smem + (hi*144) + 2*nl;      // ci = kc*2 + hi
        for (int kc = 0; kc < chunks; ++kc){
            // A fragment (weights): contiguous k = 16*hi + i, matching B
            v8h alo = *(const v8h*)(wrow + kc*32);
            v8h ahi = *(const v8h*)(wrow + kc*32 + 8);
            v16h a;
            #pragma unroll
            for (int i = 0; i < 8; ++i){ a[i] = alo[i]; a[i+8] = ahi[i]; }
            // B fragment: lane = column pixel nl, covers one ci (16 taps), branch-free
            const _Float16* bp = bbase + kc*288;             // 2 ci per chunk * 144
            v16h b;
            #pragma unroll
            for (int i = 0; i < 16; ++i) b[i] = bp[(i>>2)*36 + (i&3)];
            acc = WMMA_F16(a, b, acc);
        }
        #pragma unroll
        for (int r = 0; r < 8; ++r){
            int co = ct*16 + r + 8*hi;                 // C layout: M = r + 8*(lane>>4)
            float v = acc[r] + cb[co];
            if (mode == 1){ v = v*sc[co] + ab[co]; v = v > 0.f ? v : 0.f; }
            size_t o = (((size_t)nb*Cout + co)*OH + oy)*OW + ox0 + nl;
            if (out16) out16[o] = (_Float16)v;
            if (out32) out32[o] = v;
        }
    }
}

// =====================================================================
// Implicit-GEMM transposed conv (k=4,s=2,p=1): per output parity class,
// K = Cin*4 (2x2 taps). N-tile = 16 same-parity columns of one row.
// mode: 1 = bias+affine+relu, 2 = bias+sigmoid
// =====================================================================
__global__ void convT_k(const _Float16* __restrict__ in, const _Float16* __restrict__ wpk,
                        const float* __restrict__ cb, const float* __restrict__ sc,
                        const float* __restrict__ ab,
                        _Float16* __restrict__ out16, float* __restrict__ out32,
                        int B, int Cin, int OH, int OW, int Cout, int Cp, int mode){
    extern __shared__ _Float16 smem[];               // [Cin][2][20] (18 cols used)
    const int IH = OH >> 1, IW = OW >> 1, tpr = OW >> 4;
    int t  = blockIdx.x;
    int j  = t % tpr;
    int oy = (t / tpr) % OH;
    int nb = t / (tpr * OH);
    int px = j & 1, g = j >> 1, py = oy & 1;
    int iy0 = ((oy + py) >> 1) - 1;
    int icb = 16*g + px - 1;
    (void)B;

    int total = Cin * 2 * 18;
    for (int e = threadIdx.x; e < total; e += blockDim.x){
        int ci = e / 36; int rem = e - ci*36;
        int ry = rem / 18; int c = rem - ry*18;
        int iy = iy0 + ry, ix = icb + c;
        _Float16 v = (_Float16)0;
        if (iy >= 0 && iy < IH && ix >= 0 && ix < IW)
            v = in[(((size_t)nb*Cin + ci)*IH + iy)*IW + ix];
        smem[(ci*2 + ry)*20 + c] = v;
    }
    __syncthreads();

    int lane = threadIdx.x & 31, wv = threadIdx.x >> 5, nW = blockDim.x >> 5;
    int hi = lane >> 4, nl = lane & 15;
    int Kt = Cin*4, chunks = Kt >> 5;
    int pc = py*2 + px;

    for (int ct = wv; ct < (Cp >> 4); ct += nW){
        v8f acc = {};
        const _Float16* wrow = wpk + ((size_t)pc*Cp + ct*16 + nl)*Kt + hi*16;
        const _Float16* bbase = smem + (hi*4)*40 + nl;       // ci0 = kc*8 + hi*4, 40 = 2*20
        for (int kc = 0; kc < chunks; ++kc){
            v8h alo = *(const v8h*)(wrow + kc*32);
            v8h ahi = *(const v8h*)(wrow + kc*32 + 8);
            v16h a;
            #pragma unroll
            for (int i = 0; i < 8; ++i){ a[i] = alo[i]; a[i+8] = ahi[i]; }
            const _Float16* bp = bbase + kc*320;             // 8 ci per chunk * 40
            v16h b;
            #pragma unroll
            for (int i = 0; i < 16; ++i){
                int ci = i >> 2;
                int ty = (i >> 1) & 1, tx = i & 1;
                b[i] = bp[(ci*2 + ty)*20 + tx];
            }
            acc = WMMA_F16(a, b, acc);
        }
        #pragma unroll
        for (int r = 0; r < 8; ++r){
            int co = ct*16 + r + 8*hi;
            if (co < Cout){
                float v = acc[r] + cb[co];
                if (mode == 1){ v = v*sc[co] + ab[co]; v = v > 0.f ? v : 0.f; }
                else if (mode == 2){ v = 1.f/(1.f + __expf(-v)); }
                int ox = 32*g + px + 2*nl;
                size_t o = (((size_t)nb*Cout + co)*OH + oy)*OW + ox;
                if (out16) out16[o] = (_Float16)v;
                if (out32) out32[o] = v;
            }
        }
    }
}

// =====================================================================
// VQ helpers
// =====================================================================
__global__ void flatten_z_k(const _Float16* __restrict__ z, _Float16* __restrict__ flat,
                            float* __restrict__ fsq){
    int n = blockIdx.x*blockDim.x + threadIdx.x;
    if (n >= 8192) return;
    int b = n >> 8, gy = (n >> 4) & 15, gx = n & 15;
    float s = 0.f;
    for (int d = 0; d < 64; ++d){
        _Float16 v = z[(((size_t)b*64 + d)*16 + gy)*16 + gx];
        flat[(size_t)n*64 + d] = v;
        float f = (float)v; s += f*f;
    }
    fsq[n] = s;
}

__global__ void prep_embed_k(const float* __restrict__ e, _Float16* __restrict__ e16,
                             float* __restrict__ esq){
    int c = blockIdx.x*blockDim.x + threadIdx.x;
    if (c >= 8192) return;
    float s = 0.f;
    for (int d = 0; d < 64; ++d){
        float v = e[(size_t)c*64 + d];
        s += v*v;
        e16[(size_t)c*64 + d] = (_Float16)v;
    }
    esq[c] = s;
}

// Pass A: per 16-row tile, stream all 8192 codes via WMMA; online (min, argmin, Z)
__global__ void vq_pass_a_k(const _Float16* __restrict__ flat, const _Float16* __restrict__ e16,
                            const float* __restrict__ fsq, const float* __restrict__ esq,
                            float* __restrict__ dminO, float* __restrict__ zO,
                            int* __restrict__ codeO){
    int lane = threadIdx.x & 31, hi = lane >> 4, nl = lane & 15;
    int tn = blockIdx.x;
    const _Float16* fp = flat + ((size_t)(tn*16 + nl))*64 + hi*16;
    v8h a00 = *(const v8h*)(fp),    a01 = *(const v8h*)(fp + 8);
    v8h a10 = *(const v8h*)(fp+32), a11 = *(const v8h*)(fp + 40);
    v16h a0, a1;
    #pragma unroll
    for (int i = 0; i < 8; ++i){ a0[i]=a00[i]; a0[i+8]=a01[i]; a1[i]=a10[i]; a1[i+8]=a11[i]; }
    const float4* fq = (const float4*)(fsq + tn*16 + hi*8);
    float4 f0 = fq[0], f1 = fq[1];
    float fr[8] = {f0.x,f0.y,f0.z,f0.w,f1.x,f1.y,f1.z,f1.w};
    float m[8], Z[8]; int ar[8];
    #pragma unroll
    for (int r = 0; r < 8; ++r){ m[r] = 3.4e38f; Z[r] = 0.f; ar[r] = 0; }

    for (int ctile = 0; ctile < 512; ++ctile){
        int cN = ctile*16 + nl;
        const _Float16* ep = e16 + (size_t)cN*64 + hi*16;
        v16h b0 = *(const v16h*)(ep), b1 = *(const v16h*)(ep + 32);
        __builtin_prefetch(e16 + (size_t)(cN + 16)*64, 0, 0);   // global_prefetch_b8
        v8f acc = {};
        acc = WMMA_F16(a0, b0, acc);
        acc = WMMA_F16(a1, b1, acc);
        float eq = esq[cN];
        #pragma unroll
        for (int r = 0; r < 8; ++r){
            float d = fr[r] - 2.f*acc[r] + eq;
            if (d < m[r]){ Z[r] = Z[r]*__expf(d - m[r]) + 1.f; m[r] = d; ar[r] = cN; }
            else         { Z[r] += __expf(m[r] - d); }
        }
    }
    #pragma unroll
    for (int r = 0; r < 8; ++r){
        float mm = m[r], zz = Z[r]; int aa = ar[r];
        for (int o = 1; o < 16; o <<= 1){
            float mo = __shfl_xor(mm, o, 32);
            float zo = __shfl_xor(zz, o, 32);
            int   ao = __shfl_xor(aa, o, 32);
            if (mo < mm){ zz = zo + zz*__expf(mo - mm); mm = mo; aa = ao; }
            else        { zz = zz + zo*__expf(mm - mo); }
        }
        if (nl == 0){
            int row = tn*16 + r + 8*hi;
            dminO[row] = mm; zO[row] = zz; codeO[row] = aa;
        }
    }
}

// Pass B: per 16-code tile, stream all rows; accumulate avg softmax per code
__global__ void vq_pass_b_k(const _Float16* __restrict__ flat, const _Float16* __restrict__ e16,
                            const float* __restrict__ fsq, const float* __restrict__ esq,
                            const float* __restrict__ dmin, const float* __restrict__ Zb,
                            float* __restrict__ avgs){
    int lane = threadIdx.x & 31, hi = lane >> 4, nl = lane & 15;
    int c0 = blockIdx.x*16;
    int cN = c0 + nl;
    const _Float16* ep = e16 + (size_t)cN*64 + hi*16;
    v16h b0 = *(const v16h*)(ep), b1 = *(const v16h*)(ep + 32);
    float eq = esq[cN];
    float accS = 0.f;
    for (int tn = 0; tn < 512; ++tn){
        const _Float16* fp = flat + ((size_t)(tn*16 + nl))*64 + hi*16;
        v8h a00 = *(const v8h*)(fp),    a01 = *(const v8h*)(fp + 8);
        v8h a10 = *(const v8h*)(fp+32), a11 = *(const v8h*)(fp + 40);
        v16h a0, a1;
        #pragma unroll
        for (int i = 0; i < 8; ++i){ a0[i]=a00[i]; a0[i+8]=a01[i]; a1[i]=a10[i]; a1[i+8]=a11[i]; }
        __builtin_prefetch(flat + ((size_t)((tn+1)*16 + nl))*64, 0, 0);
        const float4* fq = (const float4*)(fsq  + tn*16 + hi*8);
        const float4* dq = (const float4*)(dmin + tn*16 + hi*8);
        const float4* zq = (const float4*)(Zb   + tn*16 + hi*8);
        float4 f0 = fq[0], f1 = fq[1], d0 = dq[0], d1 = dq[1], z0 = zq[0], z1 = zq[1];
        float fr[8] = {f0.x,f0.y,f0.z,f0.w,f1.x,f1.y,f1.z,f1.w};
        float dm[8] = {d0.x,d0.y,d0.z,d0.w,d1.x,d1.y,d1.z,d1.w};
        float zr[8] = {z0.x,z0.y,z0.z,z0.w,z1.x,z1.y,z1.z,z1.w};
        v8f acc = {};
        acc = WMMA_F16(a0, b0, acc);
        acc = WMMA_F16(a1, b1, acc);
        #pragma unroll
        for (int r = 0; r < 8; ++r){
            float d = fr[r] - 2.f*acc[r] + eq;
            accS += __expf(dm[r] - d) / zr[r];
        }
    }
    accS += __shfl_xor(accS, 16, 32);
    if (lane < 16) avgs[c0 + lane] = accS * (1.0f/8192.0f);
}

// gather zq (straight-through == zq numerically), codes, commitment accum
__global__ void vq_gather_k(const int* __restrict__ codes, const float* __restrict__ embed,
                            const _Float16* __restrict__ flat,
                            float* __restrict__ codes_out, float* __restrict__ zq_out,
                            _Float16* __restrict__ zq16, float* __restrict__ commitAcc){
    int n = blockIdx.x*blockDim.x + threadIdx.x;
    if (n >= 8192) return;
    int c = codes[n];
    codes_out[n] = (float)c;
    int b = n >> 8, gy = (n >> 4) & 15, gx = n & 15;
    float local = 0.f;
    for (int d = 0; d < 64; ++d){
        float e = embed[(size_t)c*64 + d];
        float f = (float)flat[(size_t)n*64 + d];
        float df = e - f; local += df*df;
        size_t o = (((size_t)b*64 + d)*16 + gy)*16 + gx;
        zq_out[o] = e;
        zq16[o]   = (_Float16)e;
    }
    atomicAdd(commitAcc, local);
}

__global__ void hist_k(const int* __restrict__ codes, int* __restrict__ hist){
    int n = blockIdx.x*blockDim.x + threadIdx.x;
    if (n < 8192) atomicAdd(&hist[codes[n]], 1);
}

__global__ void recon_loss_k(const float* __restrict__ recon, const float* __restrict__ x,
                             float* __restrict__ acc){
    int i = blockIdx.x*blockDim.x + threadIdx.x;
    float v = 0.f;
    if (i < 2097152){ float d = recon[i] - x[i]; v = d*d; }
    for (int o = 16; o > 0; o >>= 1) v += __shfl_down(v, o, 32);
    __shared__ float wsum[8];
    int lane = threadIdx.x & 31, w = threadIdx.x >> 5;
    if (lane == 0) wsum[w] = v;
    __syncthreads();
    if (threadIdx.x == 0){
        float s = 0.f;
        for (int k = 0; k < (int)(blockDim.x >> 5); ++k) s += wsum[k];
        atomicAdd(acc, s);
    }
}

__global__ void finalize_k(const float* __restrict__ avgs, const int* __restrict__ hist,
                           const float* __restrict__ cs, const float* __restrict__ scal,
                           float* __restrict__ out){
    __shared__ float red[256];
    int t = threadIdx.x;
    float acc = 0.f;
    for (int c = t; c < 8192; c += 256){ float a = avgs[c]; acc += a*logf(a + 1e-10f); }
    red[t] = acc; __syncthreads();
    for (int s = 128; s > 0; s >>= 1){ if (t < s) red[t] += red[t+s]; __syncthreads(); }
    float H = -red[0]; __syncthreads();

    acc = 0.f;
    for (int c = t; c < 8192; c += 256){ float p = (float)hist[c]*(1.0f/8192.0f); acc += p*logf(p + 1e-10f); }
    red[t] = acc; __syncthreads();
    for (int s = 128; s > 0; s >>= 1){ if (t < s) red[t] += red[t+s]; __syncthreads(); }
    float Hh = -red[0]; __syncthreads();

    acc = 0.f;
    for (int c = t; c < 8192; c += 256) acc += (cs[c] > 1.0f) ? 1.0f : 0.0f;
    red[t] = acc; __syncthreads();
    for (int s = 128; s > 0; s >>= 1){ if (t < s) red[t] += red[t+s]; __syncthreads(); }
    float act = red[0];

    if (t == 0){
        out[2097152] = scal[1] * (1.0f/2097152.0f);      // recon_loss
        float cl = scal[0] * (1.0f/524288.0f);
        out[2629633] = cl;                               // commitment_loss
        out[2629634] = cl;                               // codebook_loss (numerically equal)
        out[2629635] = -H / logf(8192.0f);               // diversity_loss
        out[2629636] = H;                                // usage_entropy
        out[2629637] = expf(Hh);                         // perplexity
        out[2629638] = act;                              // active_codes
    }
}

// =====================================================================
// Host side
// =====================================================================
extern "C" void kernel_launch(void* const* d_in, const int* in_sizes, int n_in,
                              void* d_out, int out_size, void* d_ws, size_t ws_size,
                              hipStream_t stream) {
    (void)in_sizes; (void)n_in; (void)out_size; (void)ws_size;
    const float* x    = (const float*)d_in[0];
    const float* ew1  = (const float*)d_in[1];
    const float* eb1  = (const float*)d_in[2];
    const float* bn1s = (const float*)d_in[3];
    const float* bn1b = (const float*)d_in[4];
    const float* ew2  = (const float*)d_in[5];
    const float* eb2  = (const float*)d_in[6];
    const float* bn2s = (const float*)d_in[7];
    const float* bn2b = (const float*)d_in[8];
    const float* ew3  = (const float*)d_in[9];
    const float* eb3  = (const float*)d_in[10];
    const float* bn3s = (const float*)d_in[11];
    const float* bn3b = (const float*)d_in[12];
    const float* ew4  = (const float*)d_in[13];
    const float* eb4  = (const float*)d_in[14];
    const float* embed= (const float*)d_in[15];
    const float* csz  = (const float*)d_in[16];
    const float* dw1  = (const float*)d_in[17];
    const float* db1  = (const float*)d_in[18];
    const float* dbn1s= (const float*)d_in[19];
    const float* dbn1b= (const float*)d_in[20];
    const float* dw2  = (const float*)d_in[21];
    const float* db2  = (const float*)d_in[22];
    const float* dbn2s= (const float*)d_in[23];
    const float* dbn2b= (const float*)d_in[24];
    const float* dw3  = (const float*)d_in[25];
    const float* db3  = (const float*)d_in[26];
    const float* dbn3s= (const float*)d_in[27];
    const float* dbn3b= (const float*)d_in[28];
    const float* dw4  = (const float*)d_in[29];
    const float* db4  = (const float*)d_in[30];
    float* out = (float*)d_out;

    char* ws = (char*)d_ws;
    size_t off = 0;
    auto alloc = [&](size_t bytes)->char*{
        char* p = ws + off; off += (bytes + 255) & ~(size_t)255; return p;
    };
    _Float16* X16  = (_Float16*)alloc((size_t)2097152*2);
    _Float16* BUFA = (_Float16*)alloc((size_t)33554432*2);   // h1 / d3
    _Float16* BUFB = (_Float16*)alloc((size_t)8388608*2);    // h2 / d2
    _Float16* BUFC = (_Float16*)alloc((size_t)4194304*2);    // h3 / d1
    _Float16* Z16  = (_Float16*)alloc((size_t)524288*2);
    _Float16* FLAT = (_Float16*)alloc((size_t)524288*2);
    _Float16* ZQ16 = (_Float16*)alloc((size_t)524288*2);
    _Float16* E16  = (_Float16*)alloc((size_t)524288*2);
    float* FSQ  = (float*)alloc(8192*4);
    float* ESQ  = (float*)alloc(8192*4);
    float* DMIN = (float*)alloc(8192*4);
    float* ZB   = (float*)alloc(8192*4);
    int*   CODE = (int*)alloc(8192*4);
    float* AVGS = (float*)alloc(8192*4);
    int*   HIST = (int*)alloc(8192*4);
    _Float16* WF1 = (_Float16*)alloc((size_t)64*32*2);
    _Float16* WF2 = (_Float16*)alloc((size_t)64*1024*2);
    _Float16* WF3 = (_Float16*)alloc((size_t)128*1024*2);
    _Float16* WF4 = (_Float16*)alloc((size_t)64*2048*2);
    _Float16* WT1 = (_Float16*)alloc((size_t)4*128*256*2);
    _Float16* WT2 = (_Float16*)alloc((size_t)4*64*512*2);
    _Float16* WT3 = (_Float16*)alloc((size_t)4*64*256*2);
    _Float16* WT4 = (_Float16*)alloc((size_t)4*16*256*2);
    float* SCAL = (float*)alloc(256);

    auto cdiv = [](int a, int b){ return (a + b - 1)/b; };

    init_k<<<cdiv(8192,256),256,0,stream>>>(HIST, SCAL);
    cvt16_k<<<cdiv(2097152,256),256,0,stream>>>(x, X16, 2097152);
    pack_fwd_w_k<<<cdiv(64*32,256),256,0,stream>>>(ew1, WF1, 64, 16, 32);
    pack_fwd_w_k<<<cdiv(64*1024,256),256,0,stream>>>(ew2, WF2, 64, 1024, 1024);
    pack_fwd_w_k<<<cdiv(128*1024,256),256,0,stream>>>(ew3, WF3, 128, 1024, 1024);
    pack_fwd_w_k<<<cdiv(64*2048,256),256,0,stream>>>(ew4, WF4, 64, 2048, 2048);
    pack_t_w_k<<<cdiv(4*128*256,256),256,0,stream>>>(dw1, WT1, 128, 64, 128);
    pack_t_w_k<<<cdiv(4*64*512,256),256,0,stream>>>(dw2, WT2, 64, 128, 64);
    pack_t_w_k<<<cdiv(4*64*256,256),256,0,stream>>>(dw3, WT3, 64, 64, 64);
    pack_t_w_k<<<cdiv(4*16*256,256),256,0,stream>>>(dw4, WT4, 1, 64, 16);

    // ---- encoder ----  (smem bytes = Cin2 * 288)
    conv_fwd_k<<<32*128*8, 128,   2*288, stream>>>(X16,  WF1, eb1, bn1s, bn1b, BUFA, nullptr, 32,  1, 256, 256,  64,   32, 1);
    conv_fwd_k<<<32*64*4,  128,  64*288, stream>>>(BUFA, WF2, eb2, bn2s, bn2b, BUFB, nullptr, 32, 64, 128, 128,  64, 1024, 1);
    conv_fwd_k<<<32*32*2,  128,  64*288, stream>>>(BUFB, WF3, eb3, bn3s, bn3b, BUFC, nullptr, 32, 64,  64,  64, 128, 1024, 1);
    conv_fwd_k<<<32*16*1,  128, 128*288, stream>>>(BUFC, WF4, eb4, nullptr, nullptr, Z16, nullptr, 32, 128, 32, 32, 64, 2048, 0);

    // ---- VQ ----
    flatten_z_k<<<32,256,0,stream>>>(Z16, FLAT, FSQ);
    prep_embed_k<<<32,256,0,stream>>>(embed, E16, ESQ);
    vq_pass_a_k<<<512,32,0,stream>>>(FLAT, E16, FSQ, ESQ, DMIN, ZB, CODE);
    vq_pass_b_k<<<512,32,0,stream>>>(FLAT, E16, FSQ, ESQ, DMIN, ZB, AVGS);
    vq_gather_k<<<32,256,0,stream>>>(CODE, embed, FLAT, out + 2097153, out + 2105345, ZQ16, SCAL);
    hist_k<<<32,256,0,stream>>>(CODE, HIST);

    // ---- decoder ----  (smem bytes = Cin * 80)
    convT_k<<<32*32*2,   128,  64*80, stream>>>(ZQ16, WT1, db1, dbn1s, dbn1b, BUFC, nullptr, 32,  64,  32,  32, 128, 128, 1);
    convT_k<<<32*64*4,   128, 128*80, stream>>>(BUFC, WT2, db2, dbn2s, dbn2b, BUFB, nullptr, 32, 128,  64,  64,  64,  64, 1);
    convT_k<<<32*128*8,  128,  64*80, stream>>>(BUFB, WT3, db3, dbn3s, dbn3b, BUFA, nullptr, 32,  64, 128, 128,  64,  64, 1);
    convT_k<<<32*256*16, 128,  64*80, stream>>>(BUFA, WT4, db4, nullptr, nullptr, nullptr, out, 32, 64, 256, 256, 1, 16, 2);

    // ---- losses / stats ----
    recon_loss_k<<<cdiv(2097152,256),256,0,stream>>>(out, x, SCAL + 1);
    finalize_k<<<1,256,0,stream>>>(AVGS, HIST, csz, SCAL, out);
}
